// NeuralAtom_59923383714417
// MI455X (gfx1250) — matrix operand order repaired and checked by reading, not compiled, not run on hardware
//
#include <hip/hip_runtime.h>
#include <math.h>

// ---------------------------------------------------------------------------
// Types
// ---------------------------------------------------------------------------
typedef _Float16 half_t;
typedef __attribute__((ext_vector_type(16))) _Float16 v16h;
typedef __attribute__((ext_vector_type(8)))  _Float16 v8h;
typedef __attribute__((ext_vector_type(8)))  float    v8f;

#define INV06     (1.0f / 0.6f)
#define INV_SQRT2 0.70710678118654752440f

// Problem constants (fixed by the reference harness)
#define BGR   32          // graphs
#define MAXN  4096        // nodes per graph
#define CEMB  256         // embedding size
#define NS    409         // valid seeds
#define SP    512         // padded seeds (multiple of 128)
#define TOTAL (BGR * MAXN)

// LDS tile strides (halfs); 40*2B = 80B row pitch keeps ds_load_b128 16B-aligned
// and spreads the 16 rows of a fragment over distinct bank groups.
#define ASTR 40
#define BSTR 40

// Epilogue modes
#define EP_NONE      0   // out = acc*scale + bias
#define EP_SSILU     1   // out = ssilu(acc + bias)
#define EP_SSILU_RES 2   // out = (res + ssilu(acc + bias)) * INV_SQRT2
#define EP_RELU_RES  4   // out = res + relu(acc + bias)
#define EP_ADD_RES   5   // out = res + acc

__device__ __forceinline__ float ssilu_f(float x) {
  return x * INV06 / (1.0f + __expf(-x));
}

// ---------------------------------------------------------------------------
// LDS-staged batched WMMA GEMM:  D = epilogue( A (MxK) * B (KxN) * scale + bias )
// A element (m,k) at A[m*sAm + k*sAk]; B element (k,n) at B[n*sBn + k*sBk].
// Template: AV => sAk==1 (A rows contiguous in k), BV => sBk==1.
//   !AV (TN) assumes sAm==1 (contiguous in m);  !BV (NN) assumes sBn==1.
// Block: 256 threads (8 waves), 128(M) x 128(N) tile; waves 4(M) x 2(N);
// wave tile 32x64 (2x4 WMMA frags). K-step 32; A/B tiles double-buffered in
// LDS so each element is fetched from memory once and served to all waves via
// ds_load_b128. Global loads for step k+32 issue before the WMMAs of step k.
// B fragments are consumed immediately to bound register pressure.
// ---------------------------------------------------------------------------
template<bool AV, bool BV>
__global__ __launch_bounds__(256)
void wmma_gemm_t(const half_t* __restrict__ A, long sAb, long sAh, int sAm, int sAk,
                 const half_t* __restrict__ B, long sBb, long sBh, int sBn, int sBk,
                 half_t* __restrict__ D, long sDb, long sDh, int ldd,
                 float* __restrict__ D32,
                 const float* __restrict__ bias,
                 const half_t* __restrict__ R, long sRb, long sRh, int ldr,
                 int M, int N, int K, int ep, float scale, int nBatchB)
{
  const int z  = blockIdx.z;
  const int bb = z % nBatchB;
  const int hh = z / nBatchB;
  A += (size_t)bb * sAb + (size_t)hh * sAh;
  B += (size_t)bb * sBb + (size_t)hh * sBh;
  if (D)   D   += (size_t)bb * sDb + (size_t)hh * sDh;
  if (D32) D32 += (size_t)bb * sDb + (size_t)hh * sDh;
  if (R)   R   += (size_t)bb * sRb + (size_t)hh * sRh;

  const int t     = threadIdx.x;
  const int lane  = t & 31;
  const int wave  = t >> 5;
  const int waveM = wave >> 1;          // 0..3
  const int waveN = wave & 1;           // 0..1
  const int m0 = blockIdx.x * 128;
  const int n0 = blockIdx.y * 128;
  const int hs = lane >> 4;             // half-wave select
  const int ml = lane & 15;

  __shared__ half_t sA[2][128 * ASTR];  // [m][k] tiles
  __shared__ half_t sB[2][128 * BSTR];  // [n][k] tiles

  // ---- global -> register tile loaders (layout aware, coalesced) ----
  // A tile: 128 rows x 32 halfs = 512 b128 chunks; 2 per thread.
  auto gloadA = [&](int k0, int i) -> v8h {
    const int idx = t + i * 256;
    if (AV) {               // rows contiguous in k: 4 b128 chunks per row
      const int row = idx >> 2, ch = idx & 3;
      return *(const v8h*)(A + (size_t)(m0 + row) * (size_t)sAm + (size_t)(k0 + ch * 8));
    } else {                // TN: contiguous in m -> read 8 m's at fixed k
      const int k = idx & 31, m = (idx >> 5) * 8;
      return *(const v8h*)(A + (size_t)(m0 + m) + (size_t)(k0 + k) * (size_t)sAk);
    }
  };
  auto storeA = [&](half_t* sa, int i, v8h v) {
    const int idx = t + i * 256;
    if (AV) {
      const int row = idx >> 2, ch = idx & 3;
      *(v8h*)&sa[row * ASTR + ch * 8] = v;
    } else {
      const int k = idx & 31, m = (idx >> 5) * 8;
#pragma unroll
      for (int j = 0; j < 8; ++j) sa[(m + j) * ASTR + k] = v[j];
    }
  };
  auto gloadB = [&](int k0, int i) -> v8h {
    const int idx = t + i * 256;
    if (BV) {               // W-layout rows contiguous in k
      const int row = idx >> 2, ch = idx & 3;
      return *(const v8h*)(B + (size_t)(n0 + row) * (size_t)sBn + (size_t)(k0 + ch * 8));
    } else {                // NN: contiguous in n -> read 8 n's at fixed k
      const int k = idx & 31, n = (idx >> 5) * 8;
      return *(const v8h*)(B + (size_t)(k0 + k) * (size_t)sBk + (size_t)(n0 + n));
    }
  };
  auto storeB = [&](half_t* sb, int i, v8h v) {
    const int idx = t + i * 256;
    if (BV) {
      const int row = idx >> 2, ch = idx & 3;
      *(v8h*)&sb[row * BSTR + ch * 8] = v;
    } else {
      const int k = idx & 31, n = (idx >> 5) * 8;
#pragma unroll
      for (int j = 0; j < 8; ++j) sb[(n + j) * BSTR + k] = v[j];
    }
  };

  // ---- LDS -> WMMA fragment readers (CDNA5 16-bit layouts, wave32) ----
  // A frag 16x32: lanes0-15: M=lane, halves0..7=K0..7, 8..15=K16..23
  //               lanes16-31: M=lane-16, halves0..7=K8..15, 8..15=K24..31
  auto fragA = [&](const half_t* sa, int mi) -> v16h {
    const int r = waveM * 32 + mi * 16 + ml;
    v8h lo = *(const v8h*)&sa[r * ASTR + 8 * hs];
    v8h hi = *(const v8h*)&sa[r * ASTR + 16 + 8 * hs];
    v16h f;
#pragma unroll
    for (int j = 0; j < 8; ++j) { f[j] = lo[j]; f[8 + j] = hi[j]; }
    return f;
  };
  // B frag 32x16: lanes0-15: N=lane, halves0..7=K0..7, 8..15=K16..23; lanes16-31: K8..15, K24..31
  auto fragB = [&](const half_t* sb, int ni) -> v16h {
    const int r = waveN * 64 + ni * 16 + ml;
    v8h lo = *(const v8h*)&sb[r * BSTR + 8 * hs];
    v8h hi = *(const v8h*)&sb[r * BSTR + 16 + 8 * hs];
    v16h f;
#pragma unroll
    for (int j = 0; j < 8; ++j) { f[j] = lo[j]; f[8 + j] = hi[j]; }
    return f;
  };

  v8f acc[2][4] = {};

  // ---- prologue: stage first K-block ----
  {
    v8h ra0 = gloadA(0, 0), ra1 = gloadA(0, 1);
    v8h rb0 = gloadB(0, 0), rb1 = gloadB(0, 1);
    storeA(sA[0], 0, ra0); storeA(sA[0], 1, ra1);
    storeB(sB[0], 0, rb0); storeB(sB[0], 1, rb1);
  }
  __syncthreads();

  int cur = 0;
  for (int k0 = 0; k0 < K; k0 += 32) {
    const bool more = (k0 + 32) < K;
    v8h ra0, ra1, rb0, rb1;
    if (more) {                       // issue next-tile loads before the math
      ra0 = gloadA(k0 + 32, 0); ra1 = gloadA(k0 + 32, 1);
      rb0 = gloadB(k0 + 32, 0); rb1 = gloadB(k0 + 32, 1);
    }
    v16h a0 = fragA(sA[cur], 0), a1 = fragA(sA[cur], 1);
#pragma unroll
    for (int ni = 0; ni < 4; ++ni) {
      v16h b = fragB(sB[cur], ni);
      acc[0][ni] = __builtin_amdgcn_wmma_f32_16x16x32_f16(false, a0, false, b, (short)0, acc[0][ni], false, false);
      acc[1][ni] = __builtin_amdgcn_wmma_f32_16x16x32_f16(false, a1, false, b, (short)0, acc[1][ni], false, false);
    }
    if (more) {
      storeA(sA[cur ^ 1], 0, ra0); storeA(sA[cur ^ 1], 1, ra1);
      storeB(sB[cur ^ 1], 0, rb0); storeB(sB[cur ^ 1], 1, rb1);
    }
    __syncthreads();
    cur ^= 1;
  }

  // ---- epilogue + store. C/D layout: VGPR r -> M = r + 8*hs, N = lane&15 ----
#pragma unroll
  for (int mi = 0; mi < 2; ++mi) {
#pragma unroll
    for (int ni = 0; ni < 4; ++ni) {
      const v8f& a = acc[mi][ni];
#pragma unroll
      for (int r = 0; r < 8; ++r) {
        const int row = m0 + waveM * 32 + mi * 16 + 8 * hs + r;
        const int col = n0 + waveN * 64 + ni * 16 + ml;
        float v = a[r] * scale;
        if (bias) v += bias[col];
        float resv = 0.0f;
        if (R) resv = (float)R[(size_t)row * (size_t)ldr + col];
        float o;
        switch (ep) {
          case EP_SSILU:     o = ssilu_f(v);                      break;
          case EP_SSILU_RES: o = (resv + ssilu_f(v)) * INV_SQRT2; break;
          case EP_RELU_RES:  o = resv + fmaxf(v, 0.0f);           break;
          case EP_ADD_RES:   o = resv + v;                        break;
          default:           o = v;                               break;
        }
        const size_t idx = (size_t)row * (size_t)ldd + col;
        if (D)   D[idx]   = (half_t)o;
        if (D32) D32[idx] = o;
      }
    }
  }
}

// ---------------------------------------------------------------------------
// Column softmax over the seed axis (reference softmaxes over dim=1 = queries).
// One thread per column; zero-fills padded seed rows [nValid, S). Coalesced.
// ---------------------------------------------------------------------------
__global__ void softmax_col_k(half_t* __restrict__ X, int S, int nValid,
                              int Ncols, long bstride)
{
  const int col = blockIdx.x * blockDim.x + threadIdx.x;
  half_t* p = X + (size_t)blockIdx.y * (size_t)bstride + col;
  float m = -1e30f;
  for (int s = 0; s < nValid; ++s) m = fmaxf(m, (float)p[(size_t)s * Ncols]);
  float sum = 0.0f;
  for (int s = 0; s < nValid; ++s) {
    float e = __expf((float)p[(size_t)s * Ncols] - m);
    sum += e;
    p[(size_t)s * Ncols] = (half_t)e;
  }
  const float inv = 1.0f / sum;
  for (int s = 0; s < nValid; ++s)
    p[(size_t)s * Ncols] = (half_t)((float)p[(size_t)s * Ncols] * inv);
  for (int s = nValid; s < S; ++s) p[(size_t)s * Ncols] = (half_t)0.0f;
}

// ---------------------------------------------------------------------------
// LayerNorm over C=256, one wave per row, in place; zeroes padded seed rows.
// ---------------------------------------------------------------------------
__global__ __launch_bounds__(256)
void layernorm_k(half_t* __restrict__ X, const float* __restrict__ g,
                 const float* __restrict__ b, int spMod, int nValid)
{
  const int warp = threadIdx.x >> 5;
  const int lane = threadIdx.x & 31;
  const long row = (long)blockIdx.x * 8 + warp;
  half_t* px = X + (size_t)row * CEMB;
  const bool zero = (spMod > 0) && ((int)(row % spMod) >= nValid);
  float v[8], s = 0.0f, s2 = 0.0f;
#pragma unroll
  for (int i = 0; i < 8; ++i) {
    v[i] = (float)px[lane + 32 * i];
    s += v[i]; s2 += v[i] * v[i];
  }
#pragma unroll
  for (int o = 16; o > 0; o >>= 1) {
    s  += __shfl_xor(s,  o, 32);
    s2 += __shfl_xor(s2, o, 32);
  }
  const float mean = s * (1.0f / CEMB);
  const float var  = s2 * (1.0f / CEMB) - mean * mean;
  const float rs = rsqrtf(var + 1e-5f);
#pragma unroll
  for (int i = 0; i < 8; ++i) {
    const int c = lane + 32 * i;
    px[c] = (half_t)(zero ? 0.0f : ((v[i] - mean) * rs * g[c] + b[c]));
  }
}

// ---------------------------------------------------------------------------
// Elementwise helpers
// ---------------------------------------------------------------------------
__global__ void cast_f32_to_f16_k(const float* __restrict__ s, half_t* __restrict__ d, long n) {
  for (long i = (long)blockIdx.x * blockDim.x + threadIdx.x; i < n;
       i += (long)gridDim.x * blockDim.x)
    d[i] = (half_t)s[i];
}
__global__ void zero_f16_k(half_t* __restrict__ d, long n) {
  for (long i = (long)blockIdx.x * blockDim.x + threadIdx.x; i < n;
       i += (long)gridDim.x * blockDim.x)
    d[i] = (half_t)0.0f;
}

// ---------------------------------------------------------------------------
// Host side
// ---------------------------------------------------------------------------
static void gemm(hipStream_t st,
                 const half_t* A, long sAb, long sAh, int sAm, int sAk,
                 const half_t* B, long sBb, long sBh, int sBn, int sBk,
                 half_t* D, long sDb, long sDh, int ldd, float* D32,
                 const float* bias,
                 const half_t* R, long sRb, long sRh, int ldr,
                 int M, int N, int K, int ep, float scale,
                 int nBatchB, int nBatch)
{
  dim3 g(M / 128, (N + 127) / 128, nBatch);
  if (sAk == 1 && sBk == 1)
    wmma_gemm_t<true, true><<<g, 256, 0, st>>>(A, sAb, sAh, sAm, sAk, B, sBb, sBh, sBn, sBk,
                                               D, sDb, sDh, ldd, D32, bias, R, sRb, sRh, ldr,
                                               M, N, K, ep, scale, nBatchB);
  else if (sAk == 1)
    wmma_gemm_t<true, false><<<g, 256, 0, st>>>(A, sAb, sAh, sAm, sAk, B, sBb, sBh, sBn, sBk,
                                                D, sDb, sDh, ldd, D32, bias, R, sRb, sRh, ldr,
                                                M, N, K, ep, scale, nBatchB);
  else
    wmma_gemm_t<false, true><<<g, 256, 0, st>>>(A, sAb, sAh, sAm, sAk, B, sBb, sBh, sBn, sBk,
                                                D, sDb, sDh, ldd, D32, bias, R, sRb, sRh, ldr,
                                                M, N, K, ep, scale, nBatchB);
}

static void cast16(hipStream_t st, const float* s, half_t* d, long n) {
  int blocks = (int)((n + 2047) / 2048); if (blocks > 4096) blocks = 4096;
  cast_f32_to_f16_k<<<blocks, 256, 0, st>>>(s, d, n);
}

extern "C" void kernel_launch(void* const* d_in, const int* in_sizes, int n_in,
                              void* d_out, int out_size, void* d_ws, size_t ws_size,
                              hipStream_t stream)
{
  (void)in_sizes; (void)n_in; (void)out_size; (void)ws_size;

  const float* h_f32 = (const float*)d_in[0];
  // d_in[1] = batch_seg: arange/MAXN -> dense batch is a pure reshape; unused.
  const float* S_f32 = (const float*)d_in[4];

  const int widx[15] = {2, 3, 5, 7, 9, 11, 17, 19, 21, 23, 29, 30, 31, 32, 33};
  const float* pbq = (const float*)d_in[6];
  const float* pbk = (const float*)d_in[8];
  const float* pbv = (const float*)d_in[10];
  const float* pbo = (const float*)d_in[12];
  const float* pg0 = (const float*)d_in[13];
  const float* pb0 = (const float*)d_in[14];
  const float* pg1 = (const float*)d_in[15];
  const float* pb1 = (const float*)d_in[16];
  const float* ibq = (const float*)d_in[18];
  const float* ibk = (const float*)d_in[20];
  const float* ibv = (const float*)d_in[22];
  const float* ibo = (const float*)d_in[24];
  const float* ig0 = (const float*)d_in[25];
  const float* ib0 = (const float*)d_in[26];
  const float* ig1 = (const float*)d_in[27];
  const float* ib1 = (const float*)d_in[28];

  // ---- workspace bump allocator (256 B aligned) ----
  char* wsb = (char*)d_ws;
  size_t off = 0;
  auto alloc = [&](size_t bytes) -> char* {
    char* p = wsb + off;
    off = (off + bytes + 255) & ~(size_t)255;
    return p;
  };
  half_t* w16[15];
  for (int i = 0; i < 15; ++i) w16[i] = (half_t*)alloc((size_t)CEMB * CEMB * 2);
  half_t* S16 = (half_t*)alloc((size_t)SP * CEMB * 2);
  half_t* BB0 = (half_t*)alloc((size_t)TOTAL * CEMB * 2);  // h16 -> V1 -> tmp
  half_t* BB1 = (half_t*)alloc((size_t)TOTAL * CEMB * 2);  // t1 -> K1 -> u1
  half_t* BB2 = (half_t*)alloc((size_t)TOTAL * CEMB * 2);  // hres -> h_back -> h_up2
  half_t* A1  = (half_t*)alloc((size_t)BGR * SP * MAXN * 2);
  half_t* Q1  = (half_t*)alloc((size_t)SP * CEMB * 2);
  half_t* P0  = (half_t*)alloc((size_t)BGR * SP * CEMB * 2);
  half_t* P1  = (half_t*)alloc((size_t)BGR * SP * CEMB * 2);
  half_t* P2  = (half_t*)alloc((size_t)BGR * SP * CEMB * 2);
  half_t* P3  = (half_t*)alloc((size_t)BGR * SP * CEMB * 2);
  half_t* P4  = (half_t*)alloc((size_t)BGR * SP * CEMB * 2);
  half_t* S2  = (half_t*)alloc((size_t)2 * BGR * SP * SP * 2);

  // Weight map: 0 Wpre1, 1 Wpre2, 2 pWq, 3 pWk, 4 pWv, 5 pWo,
  //             6 iWq, 7 iWk, 8 iWv, 9 iWo, 10 Whead, 11 Wr1a, 12 Wr1b, 13 Wr2a, 14 Wr2b

  // ---- casts ----
  for (int i = 0; i < 15; ++i)
    cast16(stream, (const float*)d_in[widx[i]], w16[i], (long)CEMB * CEMB);
  zero_f16_k<<<128, 256, 0, stream>>>(S16, (long)SP * CEMB);
  cast16(stream, S_f32, S16, (long)NS * CEMB);           // seeds, padded rows stay zero
  cast16(stream, h_f32, BB0, (long)TOTAL * CEMB);        // h16

  const float sc = 1.0f / 16.0f;   // 1/sqrt(dV), dV = 256 for both attentions

  // ---- pre residual layer: hres = (h + ssilu(ssilu(h W1^T) W2^T)) / sqrt2 ----
  gemm(stream, BB0,0,0,CEMB,1, w16[0],0,0,CEMB,1, BB1,0,0,CEMB,nullptr,
       nullptr, nullptr,0,0,0, TOTAL,CEMB,CEMB, EP_SSILU, 1.0f, 1,1);
  gemm(stream, BB1,0,0,CEMB,1, w16[1],0,0,CEMB,1, BB2,0,0,CEMB,nullptr,
       nullptr, BB0,0,0,CEMB, TOTAL,CEMB,CEMB, EP_SSILU_RES, 1.0f, 1,1);

  // ---- attn1 projections ----
  gemm(stream, S16,0,0,CEMB,1, w16[2],0,0,CEMB,1, Q1,0,0,CEMB,nullptr,
       pbq, nullptr,0,0,0, SP,CEMB,CEMB, EP_NONE, 1.0f, 1,1);
  gemm(stream, BB2,0,0,CEMB,1, w16[3],0,0,CEMB,1, BB1,0,0,CEMB,nullptr,
       pbk, nullptr,0,0,0, TOTAL,CEMB,CEMB, EP_NONE, 1.0f, 1,1);           // K1
  gemm(stream, BB2,0,0,CEMB,1, w16[4],0,0,CEMB,1, BB0,0,0,CEMB,nullptr,
       pbv, nullptr,0,0,0, TOTAL,CEMB,CEMB, EP_NONE, 1.0f, 1,1);           // V1

  // ---- attn1 scores: A1[b] = (Q1 K1[b]^T)/16, batched over 32 graphs ----
  gemm(stream, Q1,0,0,CEMB,1, BB1,(long)MAXN*CEMB,0,CEMB,1,
       A1,(long)SP*MAXN,0,MAXN,nullptr, nullptr, nullptr,0,0,0,
       SP,MAXN,CEMB, EP_NONE, sc, 32,32);
  softmax_col_k<<<dim3(MAXN/256, BGR), 256, 0, stream>>>(A1, SP, NS, MAXN, (long)SP*MAXN);

  // ---- attn1 out = Q1 + A1 @ V1 (NN), then LN, relu-FF, LN ----
  gemm(stream, A1,(long)SP*MAXN,0,MAXN,1, BB0,(long)MAXN*CEMB,0,1,CEMB,
       P0,(long)SP*CEMB,0,CEMB,nullptr, nullptr, Q1,0,0,CEMB,
       SP,CEMB,MAXN, EP_ADD_RES, 1.0f, 32,32);
  layernorm_k<<<(BGR*SP)/8, 256, 0, stream>>>(P0, pg0, pb0, SP, NS);
  gemm(stream, P0,0,0,CEMB,1, w16[5],0,0,CEMB,1, P1,0,0,CEMB,nullptr,
       pbo, P0,0,0,CEMB, BGR*SP,CEMB,CEMB, EP_RELU_RES, 1.0f, 1,1);
  layernorm_k<<<(BGR*SP)/8, 256, 0, stream>>>(P1, pg1, pb1, SP, NS);      // NAs1

  // ---- attn2 projections (2 heads of 128) ----
  gemm(stream, P1,0,0,CEMB,1, w16[6],0,0,CEMB,1, P2,0,0,CEMB,nullptr,
       ibq, nullptr,0,0,0, BGR*SP,CEMB,CEMB, EP_NONE, 1.0f, 1,1);          // Q2
  gemm(stream, P1,0,0,CEMB,1, w16[7],0,0,CEMB,1, P3,0,0,CEMB,nullptr,
       ibk, nullptr,0,0,0, BGR*SP,CEMB,CEMB, EP_NONE, 1.0f, 1,1);          // K2
  gemm(stream, P1,0,0,CEMB,1, w16[8],0,0,CEMB,1, P4,0,0,CEMB,nullptr,
       ibv, nullptr,0,0,0, BGR*SP,CEMB,CEMB, EP_NONE, 1.0f, 1,1);          // V2

  // ---- attn2 scores per (head,graph): S2[h,b] = (Q2hb K2hb^T)/16 ----
  gemm(stream, P2,(long)SP*CEMB,128,CEMB,1, P3,(long)SP*CEMB,128,CEMB,1,
       S2,(long)SP*SP,(long)BGR*SP*SP,SP,nullptr, nullptr, nullptr,0,0,0,
       SP,SP,128, EP_NONE, sc, 32,64);
  softmax_col_k<<<dim3(SP/256, 2*BGR), 256, 0, stream>>>(S2, SP, NS, SP, (long)SP*SP);

  // ---- attn2 out = Q2 + A2 @ V2 (merged heads via col offset), LN, FF, LN ----
  gemm(stream, S2,(long)SP*SP,(long)BGR*SP*SP,SP,1, P4,(long)SP*CEMB,128,1,CEMB,
       P0,(long)SP*CEMB,128,CEMB,nullptr, nullptr, P2,(long)SP*CEMB,128,CEMB,
       SP,128,SP, EP_ADD_RES, 1.0f, 32,64);
  layernorm_k<<<(BGR*SP)/8, 256, 0, stream>>>(P0, ig0, ib0, SP, NS);
  gemm(stream, P0,0,0,CEMB,1, w16[9],0,0,CEMB,1, P1,0,0,CEMB,nullptr,
       ibo, P0,0,0,CEMB, BGR*SP,CEMB,CEMB, EP_RELU_RES, 1.0f, 1,1);
  layernorm_k<<<(BGR*SP)/8, 256, 0, stream>>>(P1, ig1, ib1, SP, NS);      // NAs2

  // ---- project back: h_back[b] = A1[b]^T @ NAs2[b]  (TN GEMM) ----
  gemm(stream, A1,(long)SP*MAXN,0,1,MAXN, P1,(long)SP*CEMB,0,1,CEMB,
       BB2,(long)MAXN*CEMB,0,CEMB,nullptr, nullptr, nullptr,0,0,0,
       MAXN,CEMB,SP, EP_NONE, 1.0f, 32,32);

  // ---- head dense + two residual layers (final writes f32 to d_out) ----
  gemm(stream, BB2,0,0,CEMB,1, w16[10],0,0,CEMB,1, BB1,0,0,CEMB,nullptr,
       nullptr, nullptr,0,0,0, TOTAL,CEMB,CEMB, EP_SSILU, 1.0f, 1,1);      // u1
  gemm(stream, BB1,0,0,CEMB,1, w16[11],0,0,CEMB,1, BB0,0,0,CEMB,nullptr,
       nullptr, nullptr,0,0,0, TOTAL,CEMB,CEMB, EP_SSILU, 1.0f, 1,1);
  gemm(stream, BB0,0,0,CEMB,1, w16[12],0,0,CEMB,1, BB2,0,0,CEMB,nullptr,
       nullptr, BB1,0,0,CEMB, TOTAL,CEMB,CEMB, EP_SSILU_RES, 1.0f, 1,1);   // h_up2
  gemm(stream, BB2,0,0,CEMB,1, w16[13],0,0,CEMB,1, BB0,0,0,CEMB,nullptr,
       nullptr, nullptr,0,0,0, TOTAL,CEMB,CEMB, EP_SSILU, 1.0f, 1,1);
  gemm(stream, BB0,0,0,CEMB,1, w16[14],0,0,CEMB,1, nullptr,0,0,CEMB,(float*)d_out,
       nullptr, BB2,0,0,CEMB, TOTAL,CEMB,CEMB, EP_SSILU_RES, 1.0f, 1,1);
}
